// ParallelOPTAttention_11295763988599
// MI455X (gfx1250) — matrix operand order repaired and checked by reading, not compiled, not run on hardware
//
#include <hip/hip_runtime.h>
#include <hip/hip_bf16.h>

typedef __attribute__((ext_vector_type(16))) _Float16 v16h;
typedef __attribute__((ext_vector_type(8)))  _Float16 v8h;
typedef __attribute__((ext_vector_type(8)))  float    v8f;

#define NPAR   8
#define EMBED  1024
#define HEADS  16
#define HDIM   64
#define SEQT   1024
#define EPSV   0.01f

// ---------------------------------------------------------------------------
// CDNA5 async global->LDS copy (ASYNCcnt path). Each lane copies 16 bytes
// from its global address directly into LDS at its 32-bit LDS offset,
// without a VGPR round-trip. Completion via s_wait_asynccnt (in-order).
// ---------------------------------------------------------------------------
__device__ __forceinline__ void async_ld16(void* lds, const void* g) {
  unsigned off = (unsigned)(size_t)lds;  // flat LDS addr[31:0] == LDS offset
  asm volatile("global_load_async_to_lds_b128 %0, %1, off"
               :: "v"(off), "v"(g) : "memory");
}
__device__ __forceinline__ void wait_async_le4() {
  asm volatile("s_wait_asynccnt 0x4" ::: "memory");
}
__device__ __forceinline__ void wait_async_0() {
  asm volatile("s_wait_asynccnt 0x0" ::: "memory");
}

// ---------------------------------------------------------------------------
// WMMA fragment helpers (f16 16x16x32, wave32).
// A 16x32 (MxK): lane l holds row M = mbase + (l&15);
//   elems 0..7 -> K = kbase+(l>>4)*8+j, elems 8..15 -> +16 (two 16B reads).
// B 32x16 (KxN), LDS stored [n][k]: lane l holds col N = nbase + (l&15),
//   elems 0..7 -> K = kbase+(l>>4)*8+j, elems 8..15 -> +16 (two 16B reads).
// C/D f32: VGPR r, lane l -> M = (l>>4)*8 + r, N = l&15.
// ---------------------------------------------------------------------------
__device__ __forceinline__ v16h frag_a(const _Float16* s, int mbase, int kbase,
                                       int stride, int lane) {
  const _Float16* p = s + (mbase + (lane & 15)) * stride + kbase + (lane >> 4) * 8;
  v8h lo = *(const v8h*)p;
  v8h hi = *(const v8h*)(p + 16);
  v16h r;
#pragma unroll
  for (int j = 0; j < 8; ++j) { r[j] = lo[j]; r[j + 8] = hi[j]; }
  return r;
}

__device__ __forceinline__ v16h frag_b(const _Float16* s, int nbase, int kbase,
                                       int stride, int lane) {
  const _Float16* p = s + (nbase + (lane & 15)) * stride + kbase + (lane >> 4) * 8;
  v8h lo = *(const v8h*)p;
  v8h hi = *(const v8h*)(p + 16);
  v16h r;
#pragma unroll
  for (int j = 0; j < 8; ++j) { r[j] = lo[j]; r[j + 8] = hi[j]; }
  return r;
}

__device__ __forceinline__ v8f wmma16(v16h a, v16h b, v8f c) {
  return __builtin_amdgcn_wmma_f32_16x16x32_f16(false, a, false, b, (short)0, c,
                                                false, false);
}

__device__ __forceinline__ float xor16_max(float v) {
#pragma unroll
  for (int m = 1; m < 16; m <<= 1) v = fmaxf(v, __shfl_xor(v, m, 32));
  return v;
}
__device__ __forceinline__ float xor16_sum(float v) {
#pragma unroll
  for (int m = 1; m < 16; m <<= 1) v += __shfl_xor(v, m, 32);
  return v;
}

// ---------------------------------------------------------------------------
// W_eff[n][o][i] = W[o][i] + eps * mu[n][o][i]   (f32 -> f16)
// ---------------------------------------------------------------------------
__global__ void k_build_w(const float* __restrict__ W, const float* __restrict__ mu,
                          _Float16* __restrict__ out, int total) {
  int i = blockIdx.x * blockDim.x + threadIdx.x;
  if (i >= total) return;
  int wi = i & (EMBED * EMBED - 1);
  out[i] = (_Float16)(W[wi] + EPSV * mu[i]);
}

__global__ void k_f32_to_f16(const float* __restrict__ in, _Float16* __restrict__ out,
                             int total) {
  int i = blockIdx.x * blockDim.x + threadIdx.x;
  if (i >= total) return;
  out[i] = (_Float16)in[i];
}

// ---------------------------------------------------------------------------
// C[b][t][o] = X[b][t][:] . Weff[b][o][:] + bias[o], times scale.
// headLayout=1: f16 at [b*H + o/64][t][o%64]; headLayout=0: f32 flat [b][t][o].
// 128x128 tile, K-step 32, double-buffered LDS fed by async copies.
// ---------------------------------------------------------------------------
__global__ __launch_bounds__(256)
void k_gemm(const _Float16* __restrict__ X, const _Float16* __restrict__ Wf,
            const float* __restrict__ bias, _Float16* __restrict__ outH,
            float* __restrict__ outF, float scale, int headLayout) {
  __shared__ __align__(16) _Float16 As[2][128 * 40];
  __shared__ __align__(16) _Float16 Bs[2][128 * 40];

  int bx = blockIdx.x;
  int b  = bx >> 6;
  int t0 = ((bx >> 3) & 7) * 128;
  int o0 = (bx & 7) * 128;
  int tid  = threadIdx.x;
  int lane = tid & 31;
  int wave = tid >> 5;
  int wm = wave >> 2, wn = wave & 3;

  const _Float16* Xb = X  + ((size_t)b << 20);
  const _Float16* Wb = Wf + ((size_t)b << 20);

  v8f z = {};
  v8f acc[4][2];
#pragma unroll
  for (int im = 0; im < 4; ++im)
#pragma unroll
    for (int in = 0; in < 2; ++in) acc[im][in] = z;

  auto stage = [&](int buf, int k0) {
#pragma unroll
    for (int it = 0; it < 2; ++it) {
      int linear = tid + it * 256;
      int row = linear >> 2, c = linear & 3;
      async_ld16(&As[buf][row * 40 + c * 8],
                 &Xb[(size_t)(t0 + row) * EMBED + k0 + c * 8]);
      async_ld16(&Bs[buf][row * 40 + c * 8],
                 &Wb[(size_t)(o0 + row) * EMBED + k0 + c * 8]);
    }
  };

  stage(0, 0);
  int kb = 0;
  for (int k0 = 0; k0 < EMBED; k0 += 32, kb ^= 1) {
    if (k0 + 32 < EMBED) {
      stage(kb ^ 1, k0 + 32);   // overlap next tile copy with this tile's math
      wait_async_le4();         // in-order done => current tile landed
    } else {
      wait_async_0();
    }
    __syncthreads();

    v16h af[4], bf[2];
#pragma unroll
    for (int im = 0; im < 4; ++im) af[im] = frag_a(As[kb], wm * 64 + im * 16, 0, 40, lane);
#pragma unroll
    for (int in = 0; in < 2; ++in) bf[in] = frag_b(Bs[kb], wn * 32 + in * 16, 0, 40, lane);
#pragma unroll
    for (int im = 0; im < 4; ++im)
#pragma unroll
      for (int in = 0; in < 2; ++in) acc[im][in] = wmma16(af[im], bf[in], acc[im][in]);
    __syncthreads();  // all waves done with buf kb before it is re-filled
  }

  int half = lane >> 4, ln = lane & 15;
#pragma unroll
  for (int im = 0; im < 4; ++im)
#pragma unroll
    for (int in = 0; in < 2; ++in) {
      int gN = o0 + wn * 32 + in * 16 + ln;
      float bv = bias[gN];
#pragma unroll
      for (int r = 0; r < 8; ++r) {
        int gM = t0 + wm * 64 + im * 16 + half * 8 + r;
        float val = (acc[im][in][r] + bv) * scale;
        if (headLayout) {
          int h = gN >> 6, d = gN & 63;
          outH[((size_t)(b * HEADS + h) * SEQT + gM) * HDIM + d] = (_Float16)val;
        } else {
          outF[((size_t)b << 20) + (size_t)gM * EMBED + gN] = val;
        }
      }
    }
}

// ---------------------------------------------------------------------------
// Pass 1: per-row (max, sum exp) of S = Q K^T for one (bh, 128-row) block.
// 8 waves: wm in 0..3 (32 q-rows), wn in 0..1 (32 keys per 64-key chunk).
// ---------------------------------------------------------------------------
__global__ __launch_bounds__(256)
void k_attn_pass1(const _Float16* __restrict__ Q, const _Float16* __restrict__ K,
                  float2* __restrict__ stats) {
  __shared__ __align__(16) _Float16 Qs[128 * 72];
  __shared__ __align__(16) _Float16 Ks[64 * 72];
  __shared__ float mbuf[2][128];
  __shared__ float sbuf[2][128];

  int bh = blockIdx.x >> 3;
  int q0 = (blockIdx.x & 7) * 128;
  int tid = threadIdx.x, lane = tid & 31, wave = tid >> 5;
  int wm = wave >> 1, wn = wave & 1;

  const _Float16* Qb = Q + (size_t)bh * SEQT * HDIM;
  const _Float16* Kb = K + (size_t)bh * SEQT * HDIM;

#pragma unroll
  for (int it = 0; it < 4; ++it) {
    int linear = tid + it * 256;
    int row = linear >> 3, c = linear & 7;
    async_ld16(&Qs[row * 72 + c * 8], &Qb[(size_t)(q0 + row) * HDIM + c * 8]);
  }

  float mrun[2][8], srun[2][8];
#pragma unroll
  for (int im = 0; im < 2; ++im)
#pragma unroll
    for (int r = 0; r < 8; ++r) { mrun[im][r] = -1e30f; srun[im][r] = 0.f; }

  for (int kc = 0; kc < 16; ++kc) {
#pragma unroll
    for (int it = 0; it < 2; ++it) {
      int linear = tid + it * 256;
      int row = linear >> 3, c = linear & 7;
      async_ld16(&Ks[row * 72 + c * 8],
                 &Kb[(size_t)(kc * 64 + row) * HDIM + c * 8]);
    }
    wait_async_0();
    __syncthreads();

    v8f z = {};
    v8f acc[2][2] = {{z, z}, {z, z}};
#pragma unroll
    for (int ks = 0; ks < 2; ++ks) {
      v16h a0 = frag_a(Qs, wm * 32 + 0,  ks * 32, 72, lane);
      v16h a1 = frag_a(Qs, wm * 32 + 16, ks * 32, 72, lane);
      v16h b0 = frag_b(Ks, wn * 32 + 0,  ks * 32, 72, lane);
      v16h b1 = frag_b(Ks, wn * 32 + 16, ks * 32, 72, lane);
      acc[0][0] = wmma16(a0, b0, acc[0][0]);
      acc[0][1] = wmma16(a0, b1, acc[0][1]);
      acc[1][0] = wmma16(a1, b0, acc[1][0]);
      acc[1][1] = wmma16(a1, b1, acc[1][1]);
    }

#pragma unroll
    for (int im = 0; im < 2; ++im)
#pragma unroll
      for (int r = 0; r < 8; ++r) {
        float x0 = acc[im][0][r], x1 = acc[im][1][r];
        float mc = xor16_max(fmaxf(x0, x1));
        float sc = xor16_sum(__expf(x0 - mc) + __expf(x1 - mc));
        float mo = mrun[im][r];
        float mn = fmaxf(mo, mc);
        srun[im][r] = srun[im][r] * __expf(mo - mn) + sc * __expf(mc - mn);
        mrun[im][r] = mn;
      }
    __syncthreads();
  }

  if ((lane & 15) == 0) {
    int half = lane >> 4;
#pragma unroll
    for (int im = 0; im < 2; ++im)
#pragma unroll
      for (int r = 0; r < 8; ++r) {
        int row = wm * 32 + im * 16 + half * 8 + r;
        mbuf[wn][row] = mrun[im][r];
        sbuf[wn][row] = srun[im][r];
      }
  }
  __syncthreads();
  if (tid < 128) {
    float m0 = mbuf[0][tid], m1 = mbuf[1][tid];
    float m = fmaxf(m0, m1);
    float s = sbuf[0][tid] * __expf(m0 - m) + sbuf[1][tid] * __expf(m1 - m);
    float2 st; st.x = m; st.y = s;
    stats[(size_t)bh * SEQT + q0 + tid] = st;
  }
}

// ---------------------------------------------------------------------------
// Pass 2: double-softmax weights w = exp(exp(x-m)/s); out = (w @ V) / sum(w).
// ---------------------------------------------------------------------------
__global__ __launch_bounds__(256)
void k_attn_pass2(const _Float16* __restrict__ Q, const _Float16* __restrict__ K,
                  const _Float16* __restrict__ V, const float2* __restrict__ stats,
                  _Float16* __restrict__ attnOut) {
  __shared__ __align__(16) _Float16 Qs[128 * 72];
  __shared__ __align__(16) _Float16 Ks[64 * 72];
  __shared__ __align__(16) _Float16 Vs[64 * 72];   // [d][key]
  __shared__ __align__(16) _Float16 Ws[128 * 72];  // [qrow][key]
  __shared__ float2 sm[128];
  __shared__ float tbuf[2][128];

  int bh = blockIdx.x >> 3;
  int q0 = (blockIdx.x & 7) * 128;
  int tid = threadIdx.x, lane = tid & 31, wave = tid >> 5;
  int wm = wave >> 1, wn = wave & 1;

  const _Float16* Qb = Q + (size_t)bh * SEQT * HDIM;
  const _Float16* Kb = K + (size_t)bh * SEQT * HDIM;
  const _Float16* Vb = V + (size_t)bh * SEQT * HDIM;

#pragma unroll
  for (int it = 0; it < 4; ++it) {
    int linear = tid + it * 256;
    int row = linear >> 3, c = linear & 7;
    async_ld16(&Qs[row * 72 + c * 8], &Qb[(size_t)(q0 + row) * HDIM + c * 8]);
  }
  if (tid < 128) sm[tid] = stats[(size_t)bh * SEQT + q0 + tid];

  v8f z = {};
  v8f accO[2][2] = {{z, z}, {z, z}};
  float trun[2][8];
#pragma unroll
  for (int im = 0; im < 2; ++im)
#pragma unroll
    for (int r = 0; r < 8; ++r) trun[im][r] = 0.f;

  for (int kc = 0; kc < 16; ++kc) {
#pragma unroll
    for (int it = 0; it < 2; ++it) {
      int linear = tid + it * 256;
      int row = linear >> 3, c = linear & 7;
      async_ld16(&Ks[row * 72 + c * 8],
                 &Kb[(size_t)(kc * 64 + row) * HDIM + c * 8]);
    }
    {  // V chunk, transposed into LDS [d][key] for the B-operand layout
      int key = tid >> 2, c = tid & 3;
      v8h v0 = *(const v8h*)&Vb[(size_t)(kc * 64 + key) * HDIM + c * 16];
      v8h v1 = *(const v8h*)&Vb[(size_t)(kc * 64 + key) * HDIM + c * 16 + 8];
#pragma unroll
      for (int j = 0; j < 8; ++j) {
        Vs[(c * 16 + j) * 72 + key]     = v0[j];
        Vs[(c * 16 + 8 + j) * 72 + key] = v1[j];
      }
    }
    wait_async_0();
    __syncthreads();

    v8f acc[2][2] = {{z, z}, {z, z}};
#pragma unroll
    for (int ks = 0; ks < 2; ++ks) {
      v16h a0 = frag_a(Qs, wm * 32 + 0,  ks * 32, 72, lane);
      v16h a1 = frag_a(Qs, wm * 32 + 16, ks * 32, 72, lane);
      v16h b0 = frag_b(Ks, wn * 32 + 0,  ks * 32, 72, lane);
      v16h b1 = frag_b(Ks, wn * 32 + 16, ks * 32, 72, lane);
      acc[0][0] = wmma16(a0, b0, acc[0][0]);
      acc[0][1] = wmma16(a0, b1, acc[0][1]);
      acc[1][0] = wmma16(a1, b0, acc[1][0]);
      acc[1][1] = wmma16(a1, b1, acc[1][1]);
    }

#pragma unroll
    for (int im = 0; im < 2; ++im)
#pragma unroll
      for (int r = 0; r < 8; ++r) {
        int row = wm * 32 + im * 16 + (lane >> 4) * 8 + r;
        float2 ms = sm[row];
        float inv_s = 1.0f / ms.y;
        float w0 = __expf(__expf(acc[im][0][r] - ms.x) * inv_s);
        float w1 = __expf(__expf(acc[im][1][r] - ms.x) * inv_s);
        trun[im][r] += xor16_sum(w0 + w1);
        Ws[row * 72 + wn * 32 + (lane & 15)]      = (_Float16)w0;
        Ws[row * 72 + wn * 32 + 16 + (lane & 15)] = (_Float16)w1;
      }
    __syncthreads();

#pragma unroll
    for (int ks = 0; ks < 2; ++ks) {
      v16h a0 = frag_a(Ws, wm * 32 + 0,  ks * 32, 72, lane);
      v16h a1 = frag_a(Ws, wm * 32 + 16, ks * 32, 72, lane);
      v16h b0 = frag_b(Vs, wn * 32 + 0,  ks * 32, 72, lane);
      v16h b1 = frag_b(Vs, wn * 32 + 16, ks * 32, 72, lane);
      accO[0][0] = wmma16(a0, b0, accO[0][0]);
      accO[0][1] = wmma16(a0, b1, accO[0][1]);
      accO[1][0] = wmma16(a1, b0, accO[1][0]);
      accO[1][1] = wmma16(a1, b1, accO[1][1]);
    }
    __syncthreads();
  }

  if ((lane & 15) == 0) {
    int half = lane >> 4;
#pragma unroll
    for (int im = 0; im < 2; ++im)
#pragma unroll
      for (int r = 0; r < 8; ++r)
        tbuf[wn][wm * 32 + im * 16 + half * 8 + r] = trun[im][r];
  }
  __syncthreads();

  int b = bh >> 4, h = bh & 15;
#pragma unroll
  for (int im = 0; im < 2; ++im)
#pragma unroll
    for (int in = 0; in < 2; ++in) {
      int d = wn * 32 + in * 16 + (lane & 15);
#pragma unroll
      for (int r = 0; r < 8; ++r) {
        int row = wm * 32 + im * 16 + (lane >> 4) * 8 + r;
        float t = tbuf[0][row] + tbuf[1][row];
        float val = accO[im][in][r] / t;
        attnOut[((size_t)(b * SEQT + q0 + row)) * EMBED + h * HDIM + d] = (_Float16)val;
      }
    }
}

// ---------------------------------------------------------------------------
extern "C" void kernel_launch(void* const* d_in, const int* in_sizes, int n_in,
                              void* d_out, int out_size, void* d_ws, size_t ws_size,
                              hipStream_t stream) {
  const float* hidden = (const float*)d_in[0];
  const float* qw = (const float*)d_in[1];  const float* qb = (const float*)d_in[2];
  const float* kw = (const float*)d_in[3];  const float* kb = (const float*)d_in[4];
  const float* vw = (const float*)d_in[5];  const float* vb = (const float*)d_in[6];
  const float* ow = (const float*)d_in[7];  const float* ob = (const float*)d_in[8];
  const float* muq = (const float*)d_in[9];  const float* muk = (const float*)d_in[10];
  const float* muv = (const float*)d_in[11]; const float* muo = (const float*)d_in[12];

  char* ws = (char*)d_ws;
  const size_t MB = 1ull << 20;
  _Float16* Xh = (_Float16*)(ws + 0 * MB);
  _Float16* Wq = (_Float16*)(ws + 16 * MB);
  _Float16* Wk = (_Float16*)(ws + 32 * MB);
  _Float16* Wv = (_Float16*)(ws + 48 * MB);
  _Float16* Wo = (_Float16*)(ws + 64 * MB);
  _Float16* Qh = (_Float16*)(ws + 80 * MB);
  _Float16* Kh = (_Float16*)(ws + 96 * MB);
  _Float16* Vh = (_Float16*)(ws + 112 * MB);
  float2*  stats = (float2*)(ws + 128 * MB);
  _Float16* Ao = (_Float16*)(ws + 129 * MB);

  const int total = NPAR * EMBED * EMBED;  // 8M, same as BSZ*SEQ*EMBED
  dim3 blk(256);

  k_f32_to_f16<<<total / 256, blk, 0, stream>>>(hidden, Xh, total);
  k_build_w<<<total / 256, blk, 0, stream>>>(qw, muq, Wq, total);
  k_build_w<<<total / 256, blk, 0, stream>>>(kw, muk, Wk, total);
  k_build_w<<<total / 256, blk, 0, stream>>>(vw, muv, Wv, total);
  k_build_w<<<total / 256, blk, 0, stream>>>(ow, muo, Wo, total);

  const float scaling = 0.125f;  // HEAD_DIM^-0.5
  k_gemm<<<512, blk, 0, stream>>>(Xh, Wq, qb, Qh, nullptr, scaling, 1);
  k_gemm<<<512, blk, 0, stream>>>(Xh, Wk, kb, Kh, nullptr, 1.0f, 1);
  k_gemm<<<512, blk, 0, stream>>>(Xh, Wv, vb, Vh, nullptr, 1.0f, 1);

  k_attn_pass1<<<1024, blk, 0, stream>>>(Qh, Kh, stats);
  k_attn_pass2<<<1024, blk, 0, stream>>>(Qh, Kh, Vh, stats, Ao);

  k_gemm<<<512, blk, 0, stream>>>(Ao, Wo, ob, nullptr, (float*)d_out, 1.0f, 0);
}
